// RWKV_Block_2774548873648
// MI455X (gfx1250) — compile-verified, matrix operand order checked by reading, not run on hardware
//
#include <hip/hip_runtime.h>
#include <hip/hip_bf16.h>

#define BATCH 1024
#define EMB   2048
#define NHEAD 32
#define HSZ   64

#define BK  32
#define BKP 40   // padded LDS row stride in halves (16B aligned, bank staggered)

typedef __attribute__((ext_vector_type(16))) __bf16        v16bf;
typedef __attribute__((ext_vector_type(8)))  float         v8f;
typedef __attribute__((ext_vector_type(4)))  unsigned int  u32x4;
typedef __attribute__((ext_vector_type(8)))  unsigned int  u32x8;
typedef __attribute__((ext_vector_type(4)))  float         f32x4;

static __device__ __forceinline__ unsigned short f2bf(float f) {
    unsigned int u = __float_as_uint(f);
    u = (u + 0x7FFFu + ((u >> 16) & 1u)) >> 16;   // round-to-nearest-even
    return (unsigned short)u;
}

static __device__ __forceinline__ unsigned int pk2bf(float lo, float hi) {
#if __has_builtin(__builtin_amdgcn_cvt_pk_bf16_f32)
    typedef __attribute__((ext_vector_type(2))) __bf16 bf16x2;
    bf16x2 t = __builtin_amdgcn_cvt_pk_bf16_f32(lo, hi);
    return __builtin_bit_cast(unsigned int, t);
#else
    return (unsigned)f2bf(lo) | ((unsigned)f2bf(hi) << 16);
#endif
}

// WMMA 16-bit fragment from an LDS tile with BKP-half row stride.
// CDNA5 ISA 7.12.2: lane<16 -> row=lane, elems = K{0..7},{16..23};
// lane>=16 -> row=lane-16, elems = K{8..15},{24..31}.
static __device__ __forceinline__ v16bf frag_ld(const unsigned short* sm, int row0, int lane) {
    int l  = lane & 15;
    int hi = (lane >> 4) & 1;
    const unsigned short* p = sm + (row0 + l) * BKP + hi * 8;
    u32x4 a = *(const u32x4*)(p);
    u32x4 b = *(const u32x4*)(p + 16);
    u32x8 r;
    r[0] = a[0]; r[1] = a[1]; r[2] = a[2]; r[3] = a[3];
    r[4] = b[0]; r[5] = b[1]; r[6] = b[2]; r[7] = b[3];
    return __builtin_bit_cast(v16bf, r);
}

static __device__ __forceinline__ v8f wmma_bf16(v16bf a, v16bf b, v8f c) {
    return __builtin_amdgcn_wmma_f32_16x16x32_bf16(false, a, false, b, (short)0, c, false, false);
}

// Epilogue modes
#define EPI_NONE_F32    0
#define EPI_TANH_BF16   1
#define EPI_MIX_BF16    2   // out = bf16( p0 + p1*(pcol[n] + acc) )
#define EPI_SILU_F32    3
#define EPI_SIGMOID_F32 4
#define EPI_RELUSQ_BF16 5
#define EPI_RESID_F32   6   // out = p0 + acc
#define EPI_FINAL_F32   7   // out = p0 + p1*acc

// C[M,N] = A_bf16[M,K] @ W[N,K]^T (W fp32 or bf16, row-major over K).
// Double-buffered LDS, register prefetch, one barrier per K-step.
// Wave grid WROWS x WCOLS; each wave computes (MFR*16) x (NFR*16).
template<int TBM, int TBN, int WROWS, int WCOLS, int MFR, int NFR>
__global__ void __launch_bounds__(WROWS * WCOLS * 32) gemm_wmma(
    const unsigned short* __restrict__ A, int lda,
    const void* __restrict__ Wp, int ldw, int wf32,
    int N, int K, int epi,
    const float* __restrict__ p0, const float* __restrict__ p1,
    const float* __restrict__ pcol, void* __restrict__ outp)
{
    constexpr int THREADS = WROWS * WCOLS * 32;
    constexpr int ACH = TBM * BK / (THREADS * 16);   // 16-half chunks per thread (A)
    constexpr int BCH = TBN * BK / (THREADS * 16);   // 16-half chunks per thread (B)
    static_assert(WROWS * MFR * 16 == TBM && WCOLS * NFR * 16 == TBN, "tile mismatch");

    __shared__ __align__(16) unsigned short As[2][TBM * BKP];
    __shared__ __align__(16) unsigned short Bs[2][TBN * BKP];

    const int tid  = threadIdx.x;
    const int lane = tid & 31;
    const int w    = tid >> 5;
    const int wm   = w / WCOLS;
    const int wn   = w % WCOLS;
    const int m0   = blockIdx.y * TBM;
    const int n0   = blockIdx.x * TBN;

    v8f acc[MFR][NFR] = {};
    u32x4 pa[2 * ACH], pb[2 * BCH];

    auto fetch = [&](int k0) {
#pragma unroll
        for (int c = 0; c < ACH; ++c) {
            const int g = tid + c * THREADS;
            const int row = g >> 1, half = (g & 1) * 16;
            const unsigned short* src = A + (size_t)(m0 + row) * lda + k0 + half;
            pa[2 * c]     = *(const u32x4*)(src);
            pa[2 * c + 1] = *(const u32x4*)(src + 8);
        }
        if (wf32) {
#pragma unroll
            for (int c = 0; c < BCH; ++c) {
                const int g = tid + c * THREADS;
                const int row = g >> 1, half = (g & 1) * 16;
                const float* src = (const float*)Wp + (size_t)(n0 + row) * ldw + k0 + half;
                f32x4 f0 = *(const f32x4*)(src);
                f32x4 f1 = *(const f32x4*)(src + 4);
                f32x4 f2 = *(const f32x4*)(src + 8);
                f32x4 f3 = *(const f32x4*)(src + 12);
                u32x4 q0, q1;
                q0[0] = pk2bf(f0[0], f0[1]); q0[1] = pk2bf(f0[2], f0[3]);
                q0[2] = pk2bf(f1[0], f1[1]); q0[3] = pk2bf(f1[2], f1[3]);
                q1[0] = pk2bf(f2[0], f2[1]); q1[1] = pk2bf(f2[2], f2[3]);
                q1[2] = pk2bf(f3[0], f3[1]); q1[3] = pk2bf(f3[2], f3[3]);
                pb[2 * c] = q0; pb[2 * c + 1] = q1;
            }
        } else {
#pragma unroll
            for (int c = 0; c < BCH; ++c) {
                const int g = tid + c * THREADS;
                const int row = g >> 1, half = (g & 1) * 16;
                const unsigned short* src = (const unsigned short*)Wp + (size_t)(n0 + row) * ldw + k0 + half;
                pb[2 * c]     = *(const u32x4*)(src);
                pb[2 * c + 1] = *(const u32x4*)(src + 8);
            }
        }
    };
    auto stash = [&](int buf) {
#pragma unroll
        for (int c = 0; c < ACH; ++c) {
            const int g = tid + c * THREADS;
            const int row = g >> 1, half = (g & 1) * 16;
            *(u32x4*)(&As[buf][row * BKP + half])     = pa[2 * c];
            *(u32x4*)(&As[buf][row * BKP + half + 8]) = pa[2 * c + 1];
        }
#pragma unroll
        for (int c = 0; c < BCH; ++c) {
            const int g = tid + c * THREADS;
            const int row = g >> 1, half = (g & 1) * 16;
            *(u32x4*)(&Bs[buf][row * BKP + half])     = pb[2 * c];
            *(u32x4*)(&Bs[buf][row * BKP + half + 8]) = pb[2 * c + 1];
        }
    };

    fetch(0);
    stash(0);
    __syncthreads();
    int cur = 0;
    for (int k0 = 0; k0 < K; k0 += BK) {
        const bool more = (k0 + BK < K);
        if (more) fetch(k0 + BK);       // overlap next-tile global loads with WMMA

        v16bf af[MFR], bfr[NFR];
#pragma unroll
        for (int i = 0; i < MFR; ++i) af[i]  = frag_ld(&As[cur][0], wm * MFR * 16 + i * 16, lane);
#pragma unroll
        for (int j = 0; j < NFR; ++j) bfr[j] = frag_ld(&Bs[cur][0], wn * NFR * 16 + j * 16, lane);
#pragma unroll
        for (int i = 0; i < MFR; ++i)
#pragma unroll
            for (int j = 0; j < NFR; ++j)
                acc[i][j] = wmma_bf16(af[i], bfr[j], acc[i][j]);

        if (more) { stash(cur ^ 1); __syncthreads(); cur ^= 1; }
    }

    // ---- fused epilogue ----
    const int nc = lane & 15;
    const int hi = (lane >> 4) & 1;
#pragma unroll
    for (int i = 0; i < MFR; ++i) {
#pragma unroll
        for (int j = 0; j < NFR; ++j) {
            const int n  = n0 + wn * NFR * 16 + j * 16 + nc;
            const int mB = m0 + wm * MFR * 16 + i * 16 + hi * 8;
#pragma unroll
            for (int e = 0; e < 8; ++e) {
                const int m = mB + e;
                const size_t idx = (size_t)m * N + n;
                float a = acc[i][j][e];
                switch (epi) {
                case EPI_NONE_F32:    ((float*)outp)[idx] = a; break;
                case EPI_TANH_BF16:   ((unsigned short*)outp)[idx] = f2bf(tanhf(a)); break;
                case EPI_MIX_BF16:    ((unsigned short*)outp)[idx] =
                                          f2bf(p0[idx] + p1[idx] * (pcol[n] + a)); break;
                case EPI_SILU_F32:    ((float*)outp)[idx] = a / (1.0f + expf(-a)); break;
                case EPI_SIGMOID_F32: ((float*)outp)[idx] = 1.0f / (1.0f + expf(-a)); break;
                case EPI_RELUSQ_BF16: { float t = fmaxf(a, 0.0f);
                                        ((unsigned short*)outp)[idx] = f2bf(t * t); } break;
                case EPI_RESID_F32:   ((float*)outp)[idx] = p0[idx] + a; break;
                default:              ((float*)outp)[idx] = p0[idx] + p1[idx] * a; break;
                }
            }
        }
    }
}

// fp32 [R,C] -> bf16 [C,R]
__global__ void __launch_bounds__(256) transpose_f32_bf16(
    const float* __restrict__ in, unsigned short* __restrict__ out, int R, int C)
{
    size_t o = (size_t)blockIdx.x * 256 + threadIdx.x;
    if (o >= (size_t)R * C) return;
    int c = (int)(o / R);
    int r = (int)(o % R);
    out[o] = f2bf(in[(size_t)r * C + c]);
}

// LN1 + token shift delta + xxx (bf16); one block per batch row
__global__ void __launch_bounds__(256) ln1_kernel(
    const float* __restrict__ x, const float* __restrict__ state,
    const float* __restrict__ lw, const float* __restrict__ lb,
    const float* __restrict__ maa_x,
    float* __restrict__ xn, float* __restrict__ sx, unsigned short* __restrict__ xxx)
{
    __shared__ float sh1[8], sh2[8];
    const int b = blockIdx.x;
    const float* xr = x + (size_t)b * EMB;
    const float* st = state + (size_t)b * 66 * EMB + (size_t)1 * EMB;  // i1 = 1
    float vals[8], s = 0.f, s2 = 0.f;
#pragma unroll
    for (int i = 0; i < 8; ++i) {
        float v = xr[threadIdx.x + i * 256];
        vals[i] = v; s += v; s2 += v * v;
    }
    for (int m = 16; m >= 1; m >>= 1) { s += __shfl_xor(s, m, 32); s2 += __shfl_xor(s2, m, 32); }
    if ((threadIdx.x & 31) == 0) { sh1[threadIdx.x >> 5] = s; sh2[threadIdx.x >> 5] = s2; }
    __syncthreads();
    float ts = 0.f, ts2 = 0.f;
#pragma unroll
    for (int i = 0; i < 8; ++i) { ts += sh1[i]; ts2 += sh2[i]; }
    const float mu = ts * (1.0f / EMB);
    const float var = ts2 * (1.0f / EMB) - mu * mu;
    const float rstd = rsqrtf(var + 1e-5f);
#pragma unroll
    for (int i = 0; i < 8; ++i) {
        const int e = threadIdx.x + i * 256;
        const size_t idx = (size_t)b * EMB + e;
        float xnv = (vals[i] - mu) * rstd * lw[e] + lb[e];
        float sxv = st[e] - xnv;
        xn[idx] = xnv;
        sx[idx] = sxv;
        xxx[idx] = f2bf(xnv + sxv * maa_x[e]);
    }
}

// LN2 + ffn token shift -> xk2, xr2 (bf16)
__global__ void __launch_bounds__(256) ln2_kernel(
    const float* __restrict__ x1, const float* __restrict__ state,
    const float* __restrict__ lw, const float* __restrict__ lb,
    const float* __restrict__ mk, const float* __restrict__ mr,
    unsigned short* __restrict__ xk2, unsigned short* __restrict__ xr2)
{
    __shared__ float sh1[8], sh2[8];
    const int b = blockIdx.x;
    const float* xr = x1 + (size_t)b * EMB;
    const float* st = state + (size_t)b * 66 * EMB;   // i0 = 0
    float vals[8], s = 0.f, s2 = 0.f;
#pragma unroll
    for (int i = 0; i < 8; ++i) {
        float v = xr[threadIdx.x + i * 256];
        vals[i] = v; s += v; s2 += v * v;
    }
    for (int m = 16; m >= 1; m >>= 1) { s += __shfl_xor(s, m, 32); s2 += __shfl_xor(s2, m, 32); }
    if ((threadIdx.x & 31) == 0) { sh1[threadIdx.x >> 5] = s; sh2[threadIdx.x >> 5] = s2; }
    __syncthreads();
    float ts = 0.f, ts2 = 0.f;
#pragma unroll
    for (int i = 0; i < 8; ++i) { ts += sh1[i]; ts2 += sh2[i]; }
    const float mu = ts * (1.0f / EMB);
    const float var = ts2 * (1.0f / EMB) - mu * mu;
    const float rstd = rsqrtf(var + 1e-5f);
#pragma unroll
    for (int i = 0; i < 8; ++i) {
        const int e = threadIdx.x + i * 256;
        const size_t idx = (size_t)b * EMB + e;
        float xnv = (vals[i] - mu) * rstd * lw[e] + lb[e];
        float sxv = st[e] - xnv;
        xk2[idx] = f2bf(xnv + sxv * mk[e]);
        xr2[idx] = f2bf(xnv + sxv * mr[e]);
    }
}

static __device__ __forceinline__ float reduce64(float v, float* sh) {
    for (int m = 16; m >= 1; m >>= 1) v += __shfl_xor(v, m, 32);
    __syncthreads();
    if ((threadIdx.x & 31) == 0) sh[threadIdx.x >> 5] = v;
    __syncthreads();
    return sh[0] + sh[1];
}

// WKV readout + per-head GroupNorm + gate; one 64-thread block per (b,h)
__global__ void __launch_bounds__(64) wkv_gn_kernel(
    const float* __restrict__ r, const float* __restrict__ k, const float* __restrict__ v,
    const float* __restrict__ g, const float* __restrict__ faaaa,
    const float* __restrict__ state,
    const float* __restrict__ gnw, const float* __restrict__ gnb,
    unsigned short* __restrict__ ygg)
{
    __shared__ float rs[64];
    __shared__ float red[2];
    const int bh = blockIdx.x;
    const int b = bh >> 5;        // /NHEAD
    const int h = bh & 31;
    const int n = threadIdx.x;
    const size_t base = (size_t)b * EMB + h * HSZ;

    const float rv = r[base + n];
    const float kv = k[base + n];
    const float fv = faaaa[h * HSZ + n];
    rs[n] = rv;
    __syncthreads();

    const float c = reduce64(rv * fv * kv, red);      // sum_s r*f*k
    float y = v[base + n] * c;
    const float* st = state + (size_t)b * 66 * EMB + (size_t)2 * EMB + h * HSZ + n;
#pragma unroll 4
    for (int s = 0; s < HSZ; ++s) y += rs[s] * st[(size_t)s * EMB];

    const float s1 = reduce64(y, red);
    const float s2 = reduce64(y * y, red);
    const float mu = s1 * (1.0f / HSZ);
    const float var = s2 * (1.0f / HSZ) - mu * mu;
    const float yg = (y - mu) * rsqrtf(var + 1e-5f) * gnw[h * HSZ + n] + gnb[h * HSZ + n];
    ygg[base + n] = f2bf(yg * g[base + n]);
}

extern "C" void kernel_launch(void* const* d_in, const int* in_sizes, int n_in,
                              void* d_out, int out_size, void* d_ws, size_t ws_size,
                              hipStream_t stream)
{
    (void)in_sizes; (void)n_in; (void)out_size; (void)ws_size;
    const float* x      = (const float*)d_in[0];
    const float* state  = (const float*)d_in[1];
    const float* ln1_w  = (const float*)d_in[2];
    const float* ln1_b  = (const float*)d_in[3];
    const float* ln2_w  = (const float*)d_in[4];
    const float* ln2_b  = (const float*)d_in[5];
    const float* maa_x  = (const float*)d_in[6];
    const float* maa_k  = (const float*)d_in[8];
    const float* maa_v  = (const float*)d_in[9];
    const float* maa_r  = (const float*)d_in[10];
    const float* maa_g  = (const float*)d_in[11];
    const float* maa_w1 = (const float*)d_in[12];
    const float* maa_w2 = (const float*)d_in[13];
    // d_in[14..16]: decay path -> dead code in reference (w never used)
    const float* faaaa  = (const float*)d_in[17];
    const float* Wr     = (const float*)d_in[18];
    const float* Wk     = (const float*)d_in[19];
    const float* Wv     = (const float*)d_in[20];
    const float* Wo     = (const float*)d_in[21];
    const float* Wg     = (const float*)d_in[22];
    const float* gn_w   = (const float*)d_in[23];
    const float* gn_b   = (const float*)d_in[24];
    const float* fmk    = (const float*)d_in[25];
    const float* fmr    = (const float*)d_in[26];
    const float* fWk    = (const float*)d_in[27];
    const float* fWr    = (const float*)d_in[28];
    const float* fWv    = (const float*)d_in[29];

    const size_t P = (size_t)BATCH * EMB * 4;  // fp32 plane
    const size_t Q = (size_t)BATCH * EMB * 2;  // bf16 plane
    char* p = (char*)d_ws;
    auto alloc = [&](size_t bytes) { void* r = (void*)p; p += (bytes + 255) & ~(size_t)255; return r; };

    float* xn   = (float*)alloc(P);
    float* sx   = (float*)alloc(P);
    float* vbuf = (float*)alloc(P);
    float* gbuf = (float*)alloc(P);
    unsigned short* xxx = (unsigned short*)alloc(Q);
    unsigned short* xk  = (unsigned short*)alloc(Q);
    unsigned short* xv  = (unsigned short*)alloc(Q);
    unsigned short* xr  = (unsigned short*)alloc(Q);
    unsigned short* xg  = (unsigned short*)alloc(Q);
    unsigned short* z   = (unsigned short*)alloc((size_t)BATCH * 320 * 2);
    unsigned short* w1T = (unsigned short*)alloc((size_t)320 * EMB * 2);        // [320][2048]
    unsigned short* w2T = (unsigned short*)alloc((size_t)4 * EMB * 64 * 2);     // j=1..4 [2048][64]
    // aliases (stream-ordered reuse)
    float* rbuf  = xn;
    float* kbuf  = sx;
    float* x1    = vbuf;
    float* rrbuf = gbuf;
    unsigned short* ygg = xxx;
    unsigned short* xk2 = xk;
    unsigned short* xr2 = xv;
    unsigned short* kk  = xr;
    float* outF = (float*)d_out;

    // Variants: BIG = 64x128 block tile, 4 waves (1x4), wave 64x32 (8 WMMA / 6 frag loads)
    //           SMALL = 64x64 block tile, 4 waves (2x2), wave 32x32
    const dim3 blk(128);
    const dim3 gBig(EMB / 128, BATCH / 64);    // (16,16)

    // ---- small weight transposes (bf16) ----
    transpose_f32_bf16<<<(320 * EMB + 255) / 256, 256, 0, stream>>>(maa_w1, w1T, EMB, 320);
    for (int j = 1; j < 5; ++j)
        transpose_f32_bf16<<<(64 * EMB + 255) / 256, 256, 0, stream>>>(
            maa_w2 + (size_t)j * 64 * EMB, w2T + (size_t)(j - 1) * EMB * 64, 64, EMB);

    // ---- LN1 + shift + xxx ----
    ln1_kernel<<<BATCH, 256, 0, stream>>>(x, state, ln1_w, ln1_b, maa_x, xn, sx, xxx);

    // ---- z = tanh(xxx @ maa_w1)  [1024 x 320] ----
    gemm_wmma<64, 64, 2, 2, 2, 2><<<dim3(320 / 64, BATCH / 64), blk, 0, stream>>>(
        xxx, EMB, w1T, EMB, 0, 320, EMB, EPI_TANH_BF16, nullptr, nullptr, nullptr, z);

    // ---- mix GEMMs j=1..4 -> xk,xv,xr,xg (bf16), fused token-shift epilogue ----
    const float* maas[4] = {maa_k, maa_v, maa_r, maa_g};
    unsigned short* xouts[4] = {xk, xv, xr, xg};
    for (int j = 0; j < 4; ++j)
        gemm_wmma<64, 128, 1, 4, 4, 2><<<gBig, blk, 0, stream>>>(
            z + (j + 1) * 64, 320, w2T + (size_t)j * EMB * 64, 64, 0,
            EMB, 64, EPI_MIX_BF16, xn, sx, maas[j], xouts[j]);

    // ---- big projections ----
    gemm_wmma<64, 128, 1, 4, 4, 2><<<gBig, blk, 0, stream>>>(xr, EMB, Wr, EMB, 1, EMB, EMB, EPI_NONE_F32, nullptr, nullptr, nullptr, rbuf);
    gemm_wmma<64, 128, 1, 4, 4, 2><<<gBig, blk, 0, stream>>>(xk, EMB, Wk, EMB, 1, EMB, EMB, EPI_NONE_F32, nullptr, nullptr, nullptr, kbuf);
    gemm_wmma<64, 128, 1, 4, 4, 2><<<gBig, blk, 0, stream>>>(xv, EMB, Wv, EMB, 1, EMB, EMB, EPI_NONE_F32, nullptr, nullptr, nullptr, vbuf);
    gemm_wmma<64, 128, 1, 4, 4, 2><<<gBig, blk, 0, stream>>>(xg, EMB, Wg, EMB, 1, EMB, EMB, EPI_SILU_F32, nullptr, nullptr, nullptr, gbuf);

    // ---- WKV readout + GroupNorm + gate ----
    wkv_gn_kernel<<<BATCH * NHEAD, 64, 0, stream>>>(rbuf, kbuf, vbuf, gbuf, faaaa, state, gn_w, gn_b, ygg);

    // ---- x1 = x + (yg*g) @ Wo^T ----
    gemm_wmma<64, 128, 1, 4, 4, 2><<<gBig, blk, 0, stream>>>(ygg, EMB, Wo, EMB, 1, EMB, EMB, EPI_RESID_F32, x, nullptr, nullptr, x1);

    // ---- channel mixing ----
    ln2_kernel<<<BATCH, 256, 0, stream>>>(x1, state, ln2_w, ln2_b, fmk, fmr, xk2, xr2);
    gemm_wmma<64, 128, 1, 4, 4, 2><<<gBig, blk, 0, stream>>>(xr2, EMB, fWr, EMB, 1, EMB, EMB, EPI_SIGMOID_F32, nullptr, nullptr, nullptr, rrbuf);
    gemm_wmma<64, 128, 1, 4, 4, 2><<<gBig, blk, 0, stream>>>(xk2, EMB, fWk, EMB, 1, EMB, EMB, EPI_RELUSQ_BF16, nullptr, nullptr, nullptr, kk);
    gemm_wmma<64, 128, 1, 4, 4, 2><<<gBig, blk, 0, stream>>>(kk, EMB, fWv, EMB, 1, EMB, EMB, EPI_FINAL_F32, x1, rrbuf, nullptr, outF);
}
